// GAT_91079076479681
// MI455X (gfx1250) — compile-verified, matrix operand order checked by reading, not run on hardware
//
#include <hip/hip_runtime.h>

#define NNODES 50000
#define NEDGES 800000
#define FIN_DIM 128
#define HEADS 4
#define HC 256          // H1*HID == H2*OUT == 256
#define OUTC 64
#define EPSB 1e-5f
#define SLOPE 0.2f

typedef __attribute__((ext_vector_type(16))) _Float16 v16h;
typedef __attribute__((ext_vector_type(8)))  float    v8f;

// ---------------------------------------------------------------------------
// WMMA GEMM: C[M,NC] = A[M,K] @ B[K,NC], f32 in memory, f16 compute, f32 acc.
// K, NC compile-time -> immediate store offsets, no runtime 64-bit row math.
// Block = 256 threads (8 waves). C-tile = 64x64. Each wave: one 16-row band,
// two 16x16 col subtiles -> two v_wmma_f32_16x16x32_f16 per K-step.
// float4 staging (global_load_b128) + global_prefetch of the next K tile.
// ---------------------------------------------------------------------------
template <int K, int NC>
__launch_bounds__(256)
__global__ void gemm_wmma_f16(const float* __restrict__ A, const float* __restrict__ B,
                              float* __restrict__ C, int M) {
  __shared__ _Float16 Asm[64][32];   // [row][k]
  __shared__ _Float16 Bsm[64][32];   // [col][k]  (column-major stage)
  const int tid  = threadIdx.x;
  const int wave = tid >> 5;
  const int lane = tid & 31;
  const int rowBase = blockIdx.x * 64;
  const int colBase = blockIdx.y * 64;
  const int sr = wave & 3;     // 16-row band within tile
  const int ch = wave >> 2;    // 32-col half within tile
  const int m  = lane & 15;    // row (A) / col (B) within fragment
  const int hh = lane >> 4;    // lane half selects K sub-block per ISA layout

  v8f c0 = {}; v8f c1 = {};

  for (int kb = 0; kb < K; kb += 32) {
    // stage A: 64x32 f32 -> f16, as 512 float4 loads across 256 threads
#pragma unroll
    for (int it = 0; it < 2; ++it) {
      int idx = tid + 256 * it;
      int r = idx >> 3, f4 = (idx & 7) << 2;
      int gr = rowBase + r;
      float4 v = make_float4(0.f, 0.f, 0.f, 0.f);
      if (gr < M) {
        const float* ap = A + (long)gr * K + kb + f4;
        v = *(const float4*)ap;
        if (kb + 32 < K) __builtin_prefetch(ap + 32, 0, 3);   // global_prefetch_b8
      }
      Asm[r][f4]     = (_Float16)v.x;
      Asm[r][f4 + 1] = (_Float16)v.y;
      Asm[r][f4 + 2] = (_Float16)v.z;
      Asm[r][f4 + 3] = (_Float16)v.w;
    }
    // stage B: 32x64 f32 -> f16 transposed into [col][k]
#pragma unroll
    for (int it = 0; it < 2; ++it) {
      int idx = tid + 256 * it;
      int k = idx >> 4, c4 = (idx & 15) << 2;
      const float* bp = B + (long)(kb + k) * NC + colBase + c4;
      float4 v = *(const float4*)bp;
      if (kb + 32 < K) __builtin_prefetch(bp + 32 * NC, 0, 3);
      Bsm[c4][k]     = (_Float16)v.x;
      Bsm[c4 + 1][k] = (_Float16)v.y;
      Bsm[c4 + 2][k] = (_Float16)v.z;
      Bsm[c4 + 3][k] = (_Float16)v.w;
    }
    __syncthreads();

    v16h a, b0, b1;
#pragma unroll
    for (int j = 0; j < 8; ++j) {
      // 16-bit A 16x32 layout: VGPR j holds K = 16*(j>=4) + 8*laneHalf + 2*(j&3) (+1)
      int k0 = ((j >> 2) << 4) + (hh << 3) + ((j & 3) << 1);
      a [2 * j]     = Asm[sr * 16 + m][k0];
      a [2 * j + 1] = Asm[sr * 16 + m][k0 + 1];
      b0[2 * j]     = Bsm[ch * 32 + m][k0];
      b0[2 * j + 1] = Bsm[ch * 32 + m][k0 + 1];
      b1[2 * j]     = Bsm[ch * 32 + 16 + m][k0];
      b1[2 * j + 1] = Bsm[ch * 32 + 16 + m][k0 + 1];
    }
    c0 = __builtin_amdgcn_wmma_f32_16x16x32_f16(false, a, false, b0, (short)0, c0, false, false);
    c1 = __builtin_amdgcn_wmma_f32_16x16x32_f16(false, a, false, b1, (short)0, c1, false, false);
    __syncthreads();
  }

  // C/D layout: VGPR r -> row = r + 8*laneHalf ; col = lane&15
  const int gc   = colBase + ch * 32 + m;
  const int row0 = rowBase + sr * 16 + hh * 8;
  if (rowBase + 64 <= M) {                 // full tile: immediate-offset stores
    float* p = C + (long)row0 * NC + gc;
#pragma unroll
    for (int r = 0; r < 8; ++r) {
      p[r * NC]      = c0[r];
      p[r * NC + 16] = c1[r];
    }
  } else {
#pragma unroll
    for (int r = 0; r < 8; ++r) {
      if (row0 + r < M) {
        C[(long)(row0 + r) * NC + gc]      = c0[r];
        C[(long)(row0 + r) * NC + gc + 16] = c1[r];
      }
    }
  }
}

// ---------------------------------------------------------------------------
// Attention logits: als[n,h] = <h[n,h,:], a_src[h,:]> , same for dst.
// ---------------------------------------------------------------------------
__global__ void logits_kernel(const float* __restrict__ h, const float* __restrict__ a_src,
                              const float* __restrict__ a_dst, float* __restrict__ als,
                              float* __restrict__ ald, int n, int cc) {
  int gid = blockIdx.x * blockDim.x + threadIdx.x;
  if (gid >= n * HEADS) return;
  int node = gid / HEADS, hd = gid % HEADS;
  const float* hp = h + (long)node * HEADS * cc + hd * cc;
  const float* as = a_src + hd * cc;
  const float* ad = a_dst + hd * cc;
  float s = 0.f, d = 0.f;
  for (int c = 0; c < cc; ++c) { float v = hp[c]; s += v * as[c]; d += v * ad[c]; }
  als[gid] = s; ald[gid] = d;
}

__global__ void zero_kernel(float* __restrict__ p, long n) {
  long i = blockIdx.x * (long)blockDim.x + threadIdx.x;
  if (i < n) p[i] = 0.0f;
}

// monotonic float <-> unsigned map for atomicMax-based segment max
__device__ __forceinline__ unsigned enc_f(float f) {
  int b = __float_as_int(f);
  return (unsigned)(b >= 0 ? (b ^ 0x80000000) : ~b);
}
__device__ __forceinline__ float dec_key(unsigned k) {
  int b = (k & 0x80000000u) ? (int)(k ^ 0x80000000u) : ~(int)k;
  return __int_as_float(b);
}

__global__ void edge_max_kernel(const int* __restrict__ ei, const float* __restrict__ als,
                                const float* __restrict__ ald, unsigned* __restrict__ mkey) {
  int gid = blockIdx.x * blockDim.x + threadIdx.x;
  if (gid >= NEDGES * HEADS) return;
  int e = gid >> 2, hh = gid & 3;
  int s = ei[e], d = ei[NEDGES + e];
  float ev = als[s * HEADS + hh] + ald[d * HEADS + hh];
  ev = ev > 0.f ? ev : SLOPE * ev;
  atomicMax(&mkey[d * HEADS + hh], enc_f(ev));
}

__global__ void edge_expsum_kernel(const int* __restrict__ ei, const float* __restrict__ als,
                                   const float* __restrict__ ald, const unsigned* __restrict__ mkey,
                                   float* __restrict__ pbuf, float* __restrict__ denom) {
  int gid = blockIdx.x * blockDim.x + threadIdx.x;
  if (gid >= NEDGES * HEADS) return;
  int e = gid >> 2, hh = gid & 3;
  int s = ei[e], d = ei[NEDGES + e];
  float ev = als[s * HEADS + hh] + ald[d * HEADS + hh];
  ev = ev > 0.f ? ev : SLOPE * ev;
  float p = __expf(ev - dec_key(mkey[d * HEADS + hh]));
  pbuf[gid] = p;
  atomicAdd(&denom[d * HEADS + hh], p);
}

__global__ void edge_alpha_kernel(const int* __restrict__ ei, const float* __restrict__ denom,
                                  float* __restrict__ pbuf) {
  int gid = blockIdx.x * blockDim.x + threadIdx.x;
  if (gid >= NEDGES * HEADS) return;
  int e = gid >> 2, hh = gid & 3;
  int d = ei[NEDGES + e];
  pbuf[gid] = pbuf[gid] / (denom[d * HEADS + hh] + 1e-16f);
}

// one thread = one (edge, head, channel-pair): float2 gather + 2 atomic adds
__global__ void edge_agg_kernel(const int* __restrict__ ei, const float* __restrict__ alphab,
                                const float* __restrict__ h, float* __restrict__ acc) {
  int gid = blockIdx.x * blockDim.x + threadIdx.x;
  if (gid >= NEDGES * HEADS * 32) return;
  int c2 = gid & 31;
  int hh = (gid >> 5) & 3;
  int e  = gid >> 7;
  int s = ei[e], d = ei[NEDGES + e];
  float al = alphab[e * HEADS + hh];
  const float2 hv = *(const float2*)(h + (long)s * HC + hh * 64 + c2 * 2);
  float* dp = acc + (long)d * HC + hh * 64 + c2 * 2;
  atomicAdd(dp,     hv.x * al);
  atomicAdd(dp + 1, hv.y * al);
}

__global__ void head_mean_kernel(const float* __restrict__ acc, float* __restrict__ out, int n) {
  int gid = blockIdx.x * blockDim.x + threadIdx.x;
  if (gid >= n * OUTC) return;
  int node = gid >> 6, c = gid & 63;
  const float* p = acc + (long)node * HC + c;
  out[gid] = 0.25f * (p[0] + p[64] + p[128] + p[192]);
}

__launch_bounds__(256)
__global__ void bn_reduce_kernel(const float* __restrict__ x, float* __restrict__ sums,
                                 float* __restrict__ sqs, int n, int C) {
  int c = blockIdx.x;
  float s = 0.f, q = 0.f;
  for (int r = threadIdx.x; r < n; r += 256) {
    float v = x[(long)r * C + c];
    s += v; q += v * v;
  }
  __shared__ float sh[256], shq[256];
  sh[threadIdx.x] = s; shq[threadIdx.x] = q; __syncthreads();
  for (int o = 128; o > 0; o >>= 1) {
    if (threadIdx.x < o) { sh[threadIdx.x] += sh[threadIdx.x + o]; shq[threadIdx.x] += shq[threadIdx.x + o]; }
    __syncthreads();
  }
  if (threadIdx.x == 0) { sums[c] = sh[0]; sqs[c] = shq[0]; }
}

__global__ void bn_apply_kernel(float* __restrict__ x, const float* __restrict__ sums,
                                const float* __restrict__ sqs, const float* __restrict__ g,
                                const float* __restrict__ b, int n, int C, int relu) {
  int gid = blockIdx.x * blockDim.x + threadIdx.x;
  if (gid >= n * C) return;
  int c = gid % C;
  float mu  = sums[c] / (float)n;
  float var = sqs[c] / (float)n - mu * mu;
  float y = (x[gid] - mu) * rsqrtf(var + EPSB) * g[c] + b[c];
  if (relu) y = fmaxf(y, 0.f);
  x[gid] = y;
}

__launch_bounds__(64)
__global__ void final_kernel(const float* __restrict__ h, const float* __restrict__ w,
                             const float* __restrict__ bias, float* __restrict__ out, int n) {
  int node = blockIdx.x;
  __shared__ float sh[64];
  float v = h[(long)node * OUTC + threadIdx.x] * w[threadIdx.x];
  sh[threadIdx.x] = v; __syncthreads();
  for (int o = 32; o > 0; o >>= 1) {
    if (threadIdx.x < o) sh[threadIdx.x] += sh[threadIdx.x + o];
    __syncthreads();
  }
  if (threadIdx.x == 0) {
    float z = sh[0] + bias[0];
    out[node] = 1.f / (1.f + __expf(-z));
  }
}

// ---------------------------------------------------------------------------
extern "C" void kernel_launch(void* const* d_in, const int* in_sizes, int n_in,
                              void* d_out, int out_size, void* d_ws, size_t ws_size,
                              hipStream_t stream) {
  (void)in_sizes; (void)n_in; (void)out_size; (void)ws_size;
  const float* x      = (const float*)d_in[0];
  const int*   ei     = (const int*)d_in[1];
  const float* W1     = (const float*)d_in[2];
  const float* a1_src = (const float*)d_in[3];
  const float* a1_dst = (const float*)d_in[4];
  const float* W2     = (const float*)d_in[6];
  const float* a2_src = (const float*)d_in[7];
  const float* a2_dst = (const float*)d_in[8];
  const float* bn1_g  = (const float*)d_in[10]; const float* bn1_b = (const float*)d_in[11];
  const float* bn2_g  = (const float*)d_in[12]; const float* bn2_b = (const float*)d_in[13];
  const float* bn3_g  = (const float*)d_in[14]; const float* bn3_b = (const float*)d_in[15];
  const float* bn4_g  = (const float*)d_in[16]; const float* bn4_b = (const float*)d_in[17];
  const float* lin1_W = (const float*)d_in[18];
  const float* lin2_W = (const float*)d_in[20];
  const float* fin_W  = (const float*)d_in[22];
  const float* fin_b  = (const float*)d_in[23];
  float* out = (float*)d_out;

  // workspace carve-up (all f32)
  float* ws   = (float*)d_ws;
  float* A    = ws;                               // [N,256]  h-raw (conv1 then conv2)
  float* Bw   = A  + (long)NNODES * HC;           // [N,256]  aggregate / bn'd features
  float* C0   = Bw + (long)NNODES * HC;           // [N,64]
  float* C1   = C0 + (long)NNODES * OUTC;         // [N,64]
  float* C2   = C1 + (long)NNODES * OUTC;         // [N,64]
  float* als  = C2 + (long)NNODES * OUTC;         // [N,4]
  float* ald  = als + (long)NNODES * HEADS;       // [N,4]
  unsigned* mkey = (unsigned*)(ald + (long)NNODES * HEADS);   // [N,4]
  float* denom = (float*)mkey + (long)NNODES * HEADS;         // [N,4]
  float* pbuf  = denom + (long)NNODES * HEADS;    // [E,4]
  float* sums  = pbuf + (long)NEDGES * HEADS;     // [256]
  float* sqs   = sums + 256;                      // [256]

  const int TB = 256;
  const int nh_blk  = (NNODES * HEADS + TB - 1) / TB;
  const int eh_blk  = (NEDGES * HEADS + TB - 1) / TB;
  const int agg_blk = (NEDGES * HEADS * 32 + TB - 1) / TB;
  const int nc_blk  = ((long)NNODES * HC + TB - 1) / TB;
  const int no_blk  = (NNODES * OUTC + TB - 1) / TB;
  dim3 gemmBig((NNODES + 63) / 64, HC / 64);      // 782 x 4
  dim3 gemmLin((NNODES + 63) / 64, OUTC / 64);    // 782 x 1

  // ---- conv1 ----
  gemm_wmma_f16<FIN_DIM, HC><<<gemmBig, TB, 0, stream>>>(x, W1, A, NNODES);
  logits_kernel<<<nh_blk, TB, 0, stream>>>(A, a1_src, a1_dst, als, ald, NNODES, 64);
  zero_kernel<<<nh_blk, TB, 0, stream>>>((float*)mkey, (long)NNODES * HEADS);
  zero_kernel<<<nh_blk, TB, 0, stream>>>(denom, (long)NNODES * HEADS);
  zero_kernel<<<nc_blk, TB, 0, stream>>>(Bw, (long)NNODES * HC);
  edge_max_kernel<<<eh_blk, TB, 0, stream>>>(ei, als, ald, mkey);
  edge_expsum_kernel<<<eh_blk, TB, 0, stream>>>(ei, als, ald, mkey, pbuf, denom);
  edge_alpha_kernel<<<eh_blk, TB, 0, stream>>>(ei, denom, pbuf);
  edge_agg_kernel<<<agg_blk, TB, 0, stream>>>(ei, pbuf, A, Bw);
  bn_reduce_kernel<<<HC, TB, 0, stream>>>(Bw, sums, sqs, NNODES, HC);
  bn_apply_kernel<<<nc_blk, TB, 0, stream>>>(Bw, sums, sqs, bn1_g, bn1_b, NNODES, HC, 1);

  // ---- conv2 (reads Bw, h-raw into A, aggregate back into Bw) ----
  gemm_wmma_f16<HC, HC><<<gemmBig, TB, 0, stream>>>(Bw, W2, A, NNODES);
  logits_kernel<<<nh_blk, TB, 0, stream>>>(A, a2_src, a2_dst, als, ald, NNODES, 64);
  zero_kernel<<<nh_blk, TB, 0, stream>>>((float*)mkey, (long)NNODES * HEADS);
  zero_kernel<<<nh_blk, TB, 0, stream>>>(denom, (long)NNODES * HEADS);
  zero_kernel<<<nc_blk, TB, 0, stream>>>(Bw, (long)NNODES * HC);
  edge_max_kernel<<<eh_blk, TB, 0, stream>>>(ei, als, ald, mkey);
  edge_expsum_kernel<<<eh_blk, TB, 0, stream>>>(ei, als, ald, mkey, pbuf, denom);
  edge_alpha_kernel<<<eh_blk, TB, 0, stream>>>(ei, denom, pbuf);
  edge_agg_kernel<<<agg_blk, TB, 0, stream>>>(ei, pbuf, A, Bw);
  head_mean_kernel<<<no_blk, TB, 0, stream>>>(Bw, C0, NNODES);
  bn_reduce_kernel<<<OUTC, TB, 0, stream>>>(C0, sums, sqs, NNODES, OUTC);
  bn_apply_kernel<<<no_blk, TB, 0, stream>>>(C0, sums, sqs, bn2_g, bn2_b, NNODES, OUTC, 0);

  // ---- MLP head ----
  gemm_wmma_f16<OUTC, OUTC><<<gemmLin, TB, 0, stream>>>(C0, lin1_W, C1, NNODES);
  bn_reduce_kernel<<<OUTC, TB, 0, stream>>>(C1, sums, sqs, NNODES, OUTC);
  bn_apply_kernel<<<no_blk, TB, 0, stream>>>(C1, sums, sqs, bn3_g, bn3_b, NNODES, OUTC, 1);

  gemm_wmma_f16<OUTC, OUTC><<<gemmLin, TB, 0, stream>>>(C1, lin2_W, C2, NNODES);
  bn_reduce_kernel<<<OUTC, TB, 0, stream>>>(C2, sums, sqs, NNODES, OUTC);
  bn_apply_kernel<<<no_blk, TB, 0, stream>>>(C2, sums, sqs, bn4_g, bn4_b, NNODES, OUTC, 1);

  final_kernel<<<NNODES, 64, 0, stream>>>(C2, fin_W, fin_b, out, NNODES);
}
